// LongTermMemory_6725918785543
// MI455X (gfx1250) — matrix operand order repaired and checked by reading, not compiled, not run on hardware
//
#include <hip/hip_runtime.h>
#include <math.h>

// ---------------- CDNA5 WMMA types ----------------
typedef __attribute__((ext_vector_type(16))) __bf16 v16bf;
typedef __attribute__((ext_vector_type(8)))  float  v8f;

union BFrag {
    v16bf v;
    unsigned int u[8];
    uint4 q[2];
};

#define DIMD    128      // feature dim
#define DIMB    128      // batch (query rows)
#define CN      64       // key/value rows per chunk
#define TPB     256      // 8 wave32
#define NWAVE   8
#define KPITCH  136      // LDS pitch (ushorts) for K chunk  (272B: 16B aligned, bank-spread)
#define VPITCH  72       // LDS pitch for transposed V chunk (144B)
#define WPITCH  68       // LDS pitch for per-wave w buffer  (136B)
#define NSLICES 16
#define NBLOCKS 2048

// ---------------- bf16 helpers ----------------
__device__ __forceinline__ unsigned short f2bf(float x) {
    // RNE via hardware converter when available
    __bf16 h = (__bf16)x;
    return __builtin_bit_cast(unsigned short, h);
}
__device__ __forceinline__ float bf2f(unsigned short h) {
    return (float)__builtin_bit_cast(__bf16, h);
}
__device__ __forceinline__ unsigned int pack_bf16x2(float lo, float hi) {
#if __has_builtin(__builtin_amdgcn_cvt_pk_bf16_f32)
    auto p = __builtin_amdgcn_cvt_pk_bf16_f32(lo, hi);   // v_cvt_pk_bf16_f32
    return __builtin_bit_cast(unsigned int, p);
#else
    return (unsigned int)f2bf(lo) | ((unsigned int)f2bf(hi) << 16);
#endif
}

__device__ __forceinline__ v8f wmma_bf16(v16bf a, v16bf b, v8f c) {
    // v_wmma_f32_16x16x32_bf16  D = A*B + C
    return __builtin_amdgcn_wmma_f32_16x16x32_bf16(
        false, a, false, b, (short)0, c, false, false);
}

// B-matrix fragment (32x16 bf16): lane = N column, contiguous K within lane-half.
// p must point to (row n, k = kTile*32 + half*16), 16B aligned.
__device__ __forceinline__ v16bf load_bfrag(const unsigned short* p) {
    BFrag f;
    f.q[0] = *(const uint4*)(p);
    f.q[1] = *(const uint4*)(p + 8);
    return f.v;
}

// A-matrix fragment (16x32 bf16): lane = M row (both halves), K-pairs in VGPRs.
// VGPR g pair k = g*2 + half*16 (+ kbase). Dword (pair) gathers.
__device__ __forceinline__ v16bf load_afrag(const unsigned short* p, int pitch,
                                            int m, int kbase, int half) {
    BFrag f;
    const unsigned short* row = p + m * pitch + kbase + half * 16;
#pragma unroll
    for (int g = 0; g < 8; ++g) {
        int k = g << 1;
        f.u[g] = *(const unsigned int*)(row + k);
    }
    return f.v;
}

// ================= K0: normalize + scale + bf16 split of Q =================
__global__ void qprep_kernel(const float* __restrict__ es,
                             unsigned short* __restrict__ qh,
                             unsigned short* __restrict__ ql,
                             float invT) {
    __shared__ float red[DIMD];
    int b = blockIdx.x, t = threadIdx.x;
    float x = es[b * DIMD + t];
    red[t] = x * x;
    __syncthreads();
    for (int s = 64; s > 0; s >>= 1) {
        if (t < s) red[t] += red[t + s];
        __syncthreads();
    }
    float nrm = sqrtf(red[0]);
    float qv = x / fmaxf(nrm, 1e-12f) * invT;   // fold 1/temperature into Q
    unsigned short hi = f2bf(qv);
    qh[b * DIMD + t] = hi;
    ql[b * DIMD + t] = f2bf(qv - bf2f(hi));
}

// ================= K1: acc = sum exp(QK^T) * V ; denom partials =============
__global__ void __launch_bounds__(TPB)
pass1_kernel(const float* __restrict__ keys, const float* __restrict__ values,
             const unsigned short* __restrict__ qh,
             const unsigned short* __restrict__ ql,
             float* __restrict__ accS, float* __restrict__ denS,
             int nChunks, int cpb) {
    __shared__ __align__(16) unsigned short kh[CN * KPITCH];          // K chunk bf16
    __shared__ __align__(16) unsigned short vt[DIMD * VPITCH];        // V chunk bf16, transposed
    __shared__ __align__(16) unsigned short wb[NWAVE * 16 * WPITCH];  // per-wave softmax numerators

    const int wave = threadIdx.x >> 5, lane = threadIdx.x & 31;
    const int m0 = wave * 16, half = lane >> 4, lm = lane & 15;

    // Q fragments (resident in VGPRs for the whole kernel)
    v16bf qhf[4], qlf[4];
#pragma unroll
    for (int t = 0; t < 4; ++t) {
        qhf[t] = load_afrag(qh, DIMD, m0 + lm, t * 32, half);
        qlf[t] = load_afrag(ql, DIMD, m0 + lm, t * 32, half);
    }

    v8f vacc[8];
#pragma unroll
    for (int d = 0; d < 8; ++d) vacc[d] = (v8f){0, 0, 0, 0, 0, 0, 0, 0};
    float rowsum[8] = {0, 0, 0, 0, 0, 0, 0, 0};

    unsigned short* wrow = wb + wave * 16 * WPITCH;

    for (int c = 0; c < cpb; ++c) {
        int chunk = blockIdx.x * cpb + c;
        if (chunk >= nChunks) break;
        size_t nb = (size_t)chunk * CN;

        // ---- stage fp32 -> bf16 into LDS (vectorized: float4 in, b64 out) ----
        for (int i = threadIdx.x; i < (CN * DIMD) / 4; i += TPB) {
            int idx = i * 4;
            int n = idx >> 7, k = idx & 127;          // k multiple of 4
            float4 kv = *(const float4*)(keys   + nb * DIMD + idx);
            float4 vv = *(const float4*)(values + nb * DIMD + idx);
            uint2 kp;
            kp.x = pack_bf16x2(kv.x, kv.y);
            kp.y = pack_bf16x2(kv.z, kv.w);
            *(uint2*)(kh + n * KPITCH + k) = kp;      // ds_store_b64
            vt[(k + 0) * VPITCH + n] = f2bf(vv.x);    // transposed, per-element
            vt[(k + 1) * VPITCH + n] = f2bf(vv.y);
            vt[(k + 2) * VPITCH + n] = f2bf(vv.z);
            vt[(k + 3) * VPITCH + n] = f2bf(vv.w);
        }
        // prefetch next chunk into near caches (WGP scope): 256 thr * 128B = 32KB
        if (chunk + 1 < nChunks) {
            const char* pk = (const char*)(keys + (nb + CN) * DIMD);
            const char* pv = (const char*)(values + (nb + CN) * DIMD);
            __builtin_prefetch(pk + (size_t)threadIdx.x * 128, 0, 3);
            __builtin_prefetch(pv + (size_t)threadIdx.x * 128, 0, 3);
        }
        __syncthreads();

        // ---- S = (Qhi + Qlo) * K^T ; w = exp(S) ----
#pragma unroll
        for (int j = 0; j < 4; ++j) {
            v8f s = (v8f){0, 0, 0, 0, 0, 0, 0, 0};
            const unsigned short* krow = kh + (j * 16 + lm) * KPITCH + half * 16;
#pragma unroll
            for (int t = 0; t < 4; ++t) {
                v16bf kb = load_bfrag(krow + t * 32);
                s = wmma_bf16(qhf[t], kb, s);
                s = wmma_bf16(qlf[t], kb, s);
            }
#pragma unroll
            for (int r = 0; r < 8; ++r) {
                float w = __expf(s[r]);       // |s|<=20.1 -> no overflow
                rowsum[r] += w;
                wrow[(r + half * 8) * WPITCH + j * 16 + lm] = f2bf(w);
            }
        }

        // ---- vacc += w * V  (w re-read from LDS in A-fragment layout) ----
        v16bf wf[2];
#pragma unroll
        for (int t = 0; t < 2; ++t) wf[t] = load_afrag(wrow, WPITCH, lm, t * 32, half);
#pragma unroll
        for (int d = 0; d < 8; ++d) {
            const unsigned short* vrow = vt + (d * 16 + lm) * VPITCH + half * 16;
#pragma unroll
            for (int t = 0; t < 2; ++t) {
                v16bf vb = load_bfrag(vrow + t * 32);
                vacc[d] = wmma_bf16(wf[t], vb, vacc[d]);
            }
        }
        __syncthreads();
    }

    // ---- flush accumulators (sliced to reduce atomic contention) ----
    const int slice = blockIdx.x & (NSLICES - 1);
    float* acc = accS + slice * DIMB * DIMD;
    float* den = denS + slice * DIMB;
#pragma unroll
    for (int d = 0; d < 8; ++d)
#pragma unroll
        for (int r = 0; r < 8; ++r)
            atomicAdd(&acc[(m0 + r + half * 8) * DIMD + d * 16 + lm], vacc[d][r]);
#pragma unroll
    for (int r = 0; r < 8; ++r) {
        float v = rowsum[r];
        v += __shfl_xor(v, 1, 32);
        v += __shfl_xor(v, 2, 32);
        v += __shfl_xor(v, 4, 32);
        v += __shfl_xor(v, 8, 32);
        if (lm == 0) atomicAdd(&den[m0 + r + half * 8], v);
    }
}

// ================= K2: out = acc/denom ; lnd = log(denom) ==================
__global__ void finalize_kernel(const float* __restrict__ accS,
                                const float* __restrict__ denS,
                                float* __restrict__ out,
                                float* __restrict__ lnd) {
    int b = blockIdx.x, t = threadIdx.x;
    float den = 0.f, a = 0.f;
#pragma unroll
    for (int s = 0; s < NSLICES; ++s) {
        den += denS[s * DIMB + b];
        a += accS[s * DIMB * DIMD + b * DIMD + t];
    }
    out[b * DIMD + t] = a / den;
    if (t == 0) lnd[b] = __logf(den);
}

// ================= K3: new_age[n] = age[n] + sum_b exp(s - ln d[b]) =========
__global__ void __launch_bounds__(TPB)
pass2_kernel(const float* __restrict__ keys, const float* __restrict__ age,
             const unsigned short* __restrict__ qh,
             const unsigned short* __restrict__ ql,
             const float* __restrict__ lnd,
             float* __restrict__ outAge, int nChunks, int cpb) {
    __shared__ __align__(16) unsigned short kh[CN * KPITCH];
    __shared__ float wpart[NWAVE][CN];

    const int wave = threadIdx.x >> 5, lane = threadIdx.x & 31;
    const int m0 = wave * 16, half = lane >> 4, lm = lane & 15;

    v16bf qhf[4], qlf[4];
#pragma unroll
    for (int t = 0; t < 4; ++t) {
        qhf[t] = load_afrag(qh, DIMD, m0 + lm, t * 32, half);
        qlf[t] = load_afrag(ql, DIMD, m0 + lm, t * 32, half);
    }
    float lr[8];
#pragma unroll
    for (int r = 0; r < 8; ++r) lr[r] = lnd[m0 + r + half * 8];

    for (int c = 0; c < cpb; ++c) {
        int chunk = blockIdx.x * cpb + c;
        if (chunk >= nChunks) break;
        size_t nb = (size_t)chunk * CN;

        for (int i = threadIdx.x; i < (CN * DIMD) / 4; i += TPB) {
            int idx = i * 4;
            int n = idx >> 7, k = idx & 127;
            float4 kv = *(const float4*)(keys + nb * DIMD + idx);
            uint2 kp;
            kp.x = pack_bf16x2(kv.x, kv.y);
            kp.y = pack_bf16x2(kv.z, kv.w);
            *(uint2*)(kh + n * KPITCH + k) = kp;
        }
        if (chunk + 1 < nChunks) {
            const char* pk = (const char*)(keys + (nb + CN) * DIMD);
            __builtin_prefetch(pk + (size_t)threadIdx.x * 128, 0, 3);
        }
        __syncthreads();

#pragma unroll
        for (int j = 0; j < 4; ++j) {
            v8f s = (v8f){0, 0, 0, 0, 0, 0, 0, 0};
            const unsigned short* krow = kh + (j * 16 + lm) * KPITCH + half * 16;
#pragma unroll
            for (int t = 0; t < 4; ++t) {
                v16bf kb = load_bfrag(krow + t * 32);
                s = wmma_bf16(qhf[t], kb, s);
                s = wmma_bf16(qlf[t], kb, s);
            }
            float p = 0.f;
#pragma unroll
            for (int r = 0; r < 8; ++r) p += __expf(s[r] - lr[r]);
            p += __shfl_xor(p, 16, 32);           // add other half's 8 rows
            if (half == 0) wpart[wave][j * 16 + lm] = p;
        }
        __syncthreads();

        for (int n = threadIdx.x; n < CN; n += TPB) {
            float sum = 0.f;
#pragma unroll
            for (int w = 0; w < NWAVE; ++w) sum += wpart[w][n];
            outAge[nb + n] = age[nb + n] + sum;
        }
        // next iteration's __syncthreads (after staging) protects wpart reuse
    }
}

// =========================== host launcher =================================
extern "C" void kernel_launch(void* const* d_in, const int* in_sizes, int n_in,
                              void* d_out, int out_size, void* d_ws, size_t ws_size,
                              hipStream_t stream) {
    const float* es     = (const float*)d_in[0];
    const float* keys   = (const float*)d_in[1];
    const float* values = (const float*)d_in[2];
    const float* age    = (const float*)d_in[3];

    const int N = in_sizes[3];                 // number of memory rows (1M)
    const float temp = 0.11f - log10f((float)N) * 0.01f;
    const float invT = 1.0f / temp;

    // workspace layout
    unsigned short* qh = (unsigned short*)d_ws;                  // 128*128 bf16
    unsigned short* ql = qh + DIMB * DIMD;                       // 128*128 bf16
    float* fws  = (float*)((char*)d_ws + 2 * DIMB * DIMD * sizeof(unsigned short));
    float* lnd  = fws;                                           // 128
    float* denS = fws + DIMB;                                    // 16*128
    float* accS = denS + NSLICES * DIMB;                         // 16*128*128

    float* out    = (float*)d_out;          // weighted_sum [128,128]
    float* outAge = out + DIMB * DIMD;      // new_age [N]

    const int nChunks = N / CN;
    const int cpb = (nChunks + NBLOCKS - 1) / NBLOCKS;

    // zero accumulator slices (denS + accS are contiguous)
    hipMemsetAsync(denS, 0, (size_t)(NSLICES * DIMB + NSLICES * DIMB * DIMD) * sizeof(float),
                   stream);

    qprep_kernel<<<DIMB, DIMD, 0, stream>>>(es, qh, ql, invT);
    pass1_kernel<<<NBLOCKS, TPB, 0, stream>>>(keys, values, qh, ql, accS, denS,
                                              nChunks, cpb);
    finalize_kernel<<<DIMB, DIMD, 0, stream>>>(accS, denS, out, lnd);
    pass2_kernel<<<NBLOCKS, TPB, 0, stream>>>(keys, age, qh, ql, lnd, outAge,
                                              nChunks, cpb);
}
